// Projector_18219251270113
// MI455X (gfx1250) — matrix operand order repaired and checked by reading, not compiled, not run on hardware
//
#include <hip/hip_runtime.h>
#include <hip/hip_bf16.h>
#include <math.h>

// ---------------------------------------------------------------------------
// Projector: out = feature + attention(feature-pixels as Q; conv1x1(token) as K,V)
// N=8, C=512, H=W=128 (HW=16384), heads=8, d=64, L=64, CT=256
// Memory-bound (~537 MB @ 23.3 TB/s ~ 23us); GEMMs (17 GFLOP) via f16 WMMA.
// ---------------------------------------------------------------------------

typedef __attribute__((ext_vector_type(16))) _Float16 v16h;
typedef __attribute__((ext_vector_type(8)))  _Float16 v8h;
typedef __attribute__((ext_vector_type(8)))  float    v8f;

union ABOp { v16h v; v8h h[2]; };

#define NB      8
#define CCH     512
#define HEADS   8
#define DHEAD   64
#define LTOK    64
#define CT      256
#define HWPIX   16384
#define PIXB    256      // pixels per block
#define QS      72       // LDS stride (halfs) for Q tile   (mult of 8 -> 16B aligned rows)
#define KS      72       // LDS stride (halfs) for K^T
#define VS      72       // LDS stride (halfs) for V
#define LSF     68       // LDS stride (floats) for logits  (row = 272 B, 16B aligned)
#define CSH     136      // same buffer viewed as halfs for coef rows

// ---------------------------------------------------------------------------
// Kernel 1: token projections.  v = Wv@token + bv ; k = Wk@token + bk
// Writes f16:  V as [nh][dd][l]  (A-operand of GEMM2, contraction l contiguous)
//              K^T as [nh][l][dd] (B-operand of GEMM1, contraction dd contiguous)
// ---------------------------------------------------------------------------
__global__ __launch_bounds__(64)
void proj_kv_kernel(const float* __restrict__ token,
                    const float* __restrict__ Wv, const float* __restrict__ bv,
                    const float* __restrict__ Wk, const float* __restrict__ bk,
                    _Float16* __restrict__ vout,  // [64][64][64]
                    _Float16* __restrict__ ktout) // [64][64][64]
{
    const int nc = blockIdx.x;        // 0..N*C-1
    const int n  = nc >> 9;           // /512
    const int c  = nc & (CCH - 1);
    const int l  = threadIdx.x;       // 0..63

    const float* tok = token + (size_t)n * CT * LTOK + l;
    const float* wv  = Wv + (size_t)c * CT;   // uniform across lanes -> scalar loads
    const float* wk  = Wk + (size_t)c * CT;

    float av = 0.f, ak = 0.f;
    #pragma unroll 4
    for (int ct = 0; ct < CT; ++ct) {
        float t = tok[(size_t)ct * LTOK];
        av = fmaf(wv[ct], t, av);
        ak = fmaf(wk[ct], t, ak);
    }
    av += bv[c];
    ak += bk[c];

    const int hh = c >> 6, dd = c & 63;
    const int nh = n * HEADS + hh;
    vout [((size_t)nh * DHEAD + dd) * LTOK + l ] = (_Float16)av;
    ktout[((size_t)nh * LTOK  + l ) * DHEAD + dd] = (_Float16)ak;
}

// ---------------------------------------------------------------------------
// Kernel 2: per (n,h), per 256-pixel tile:
//   logits = (Q*K)/8 ; coef = softmax_L(logits) ; out = feature + V*coef^T
// ---------------------------------------------------------------------------
__global__ __launch_bounds__(256)
void attn_main_kernel(const float* __restrict__ feature,
                      const _Float16* __restrict__ ktg,  // [nh][l][dd]
                      const _Float16* __restrict__ vg_,  // [nh][dd][l]
                      float* __restrict__ out)
{
    __shared__ _Float16 qs[PIXB * QS];     // Q tile, f16  (36864 B)
    __shared__ _Float16 ks[LTOK * KS];     // K^T,    f16  ( 9216 B)
    __shared__ _Float16 vs[DHEAD * VS];    // V,      f16  ( 9216 B)
    __shared__ float    lg[PIXB * LSF];    // logits f32, reused as coef f16 (69632 B)

    const int nh   = blockIdx.y;                 // n*8 + head; channel base = nh*64
    const int pix0 = blockIdx.x * PIXB;
    const int tid  = threadIdx.x;
    const int lid  = tid & 31;                   // wave32 lane
    const int wv32 = tid >> 5;                   // wave id 0..7
    const int half = lid >> 4;                   // lane-half select
    const int ln   = lid & 15;
    const int wrow = wv32 * 32;                  // 32 rows (pixels) per wave

    // ---- stage K^T and V into LDS ----
    {
        const _Float16* kg = ktg + (size_t)nh * (LTOK * DHEAD);
        const _Float16* vgp = vg_ + (size_t)nh * (DHEAD * LTOK);
        for (int i = tid; i < LTOK * DHEAD; i += 256) {
            int r = i >> 6, c = i & 63;
            ks[r * KS + c] = kg[i];
            vs[r * VS + c] = vgp[i];
        }
    }
    // ---- stage Q tile (feature -> f16), transposed to [pixel][dd] ----
    {
        const float* fbase = feature + (size_t)nh * DHEAD * HWPIX + pix0;
        #pragma unroll 4
        for (int dd = 0; dd < DHEAD; ++dd) {
            float x = fbase[(size_t)dd * HWPIX + tid];   // coalesced per row
            qs[tid * QS + dd] = (_Float16)x;
        }
    }
    __syncthreads();

    // ---- GEMM1: logits[32 x 64] per wave = Q(32x64) * K(64x64) ----
    v8f acc[2][4];
    #pragma unroll
    for (int mt = 0; mt < 2; ++mt)
        #pragma unroll
        for (int j = 0; j < 4; ++j) acc[mt][j] = (v8f){0.f,0.f,0.f,0.f,0.f,0.f,0.f,0.f};

    #pragma unroll
    for (int s = 0; s < 2; ++s) {                // K-dim steps of 32 over d
        ABOp a[2];
        #pragma unroll
        for (int mt = 0; mt < 2; ++mt) {
            const int m = wrow + mt * 16 + ln;
            a[mt].h[0] = *(const v8h*)&qs[m * QS + s * 32 + half * 8];
            a[mt].h[1] = *(const v8h*)&qs[m * QS + s * 32 + 16 + half * 8];
        }
        #pragma unroll
        for (int j = 0; j < 4; ++j) {            // N-tiles over L
            ABOp b;
            const int col = j * 16 + ln;         // l column
            b.h[0] = *(const v8h*)&ks[col * KS + s * 32 + half * 8];
            b.h[1] = *(const v8h*)&ks[col * KS + s * 32 + 16 + half * 8];
            #pragma unroll
            for (int mt = 0; mt < 2; ++mt)
                acc[mt][j] = __builtin_amdgcn_wmma_f32_16x16x32_f16(
                    false, a[mt].v, false, b.v, (short)0, acc[mt][j], false, false);
        }
    }

    // ---- spill logits*(1/sqrt(d)) to LDS  (C-layout: VGPR r -> row r / r+8) ----
    #pragma unroll
    for (int mt = 0; mt < 2; ++mt)
        #pragma unroll
        for (int j = 0; j < 4; ++j)
            #pragma unroll
            for (int r = 0; r < 8; ++r) {
                const int row = wrow + mt * 16 + half * 8 + r;
                const int col = j * 16 + ln;
                lg[row * LSF + col] = acc[mt][j][r] * 0.125f;
            }
    __syncthreads();

    // ---- softmax over L, one pixel-row per thread; coef f16 in-place ----
    {
        float vals[LTOK];
        float mx = -INFINITY;
        #pragma unroll
        for (int l = 0; l < LTOK; ++l) {
            vals[l] = lg[tid * LSF + l];
            mx = fmaxf(mx, vals[l]);
        }
        float ssum = 0.f;
        #pragma unroll
        for (int l = 0; l < LTOK; ++l) {
            float e = __expf(vals[l] - mx);
            vals[l] = e;
            ssum += e;
        }
        const float rs = 1.0f / ssum;
        _Float16* crow = (_Float16*)lg;          // same bytes, half view
        #pragma unroll
        for (int l = 0; l < LTOK; ++l)
            crow[tid * CSH + l] = (_Float16)(vals[l] * rs);
    }
    __syncthreads();

    // ---- GEMM2: P(64 x 32 pixels per wave) = V(64x64) * coef^T(64x32) ----
    v8f acc2[4][2];
    #pragma unroll
    for (int mm = 0; mm < 4; ++mm)
        #pragma unroll
        for (int nt = 0; nt < 2; ++nt) acc2[mm][nt] = (v8f){0.f,0.f,0.f,0.f,0.f,0.f,0.f,0.f};

    const _Float16* crow = (const _Float16*)lg;
    #pragma unroll
    for (int s = 0; s < 2; ++s) {                // K-dim steps of 32 over L
        ABOp a[4];
        #pragma unroll
        for (int mm = 0; mm < 4; ++mm) {         // M-tiles over d
            const int m = mm * 16 + ln;          // dd row
            a[mm].h[0] = *(const v8h*)&vs[m * VS + s * 32 + half * 8];
            a[mm].h[1] = *(const v8h*)&vs[m * VS + s * 32 + 16 + half * 8];
        }
        #pragma unroll
        for (int nt = 0; nt < 2; ++nt) {         // N-tiles over this wave's pixels
            const int p = wrow + nt * 16 + ln;
            ABOp b;
            b.h[0] = *(const v8h*)&crow[p * CSH + s * 32 + half * 8];
            b.h[1] = *(const v8h*)&crow[p * CSH + s * 32 + 16 + half * 8];
            #pragma unroll
            for (int mm = 0; mm < 4; ++mm)
                acc2[mm][nt] = __builtin_amdgcn_wmma_f32_16x16x32_f16(
                    false, a[mm].v, false, b.v, (short)0, acc2[mm][nt], false, false);
        }
    }

    // ---- epilogue: out = feature + proj  (feature tile is hot in L2/WGP$) ----
    const size_t base = (size_t)nh * DHEAD * HWPIX + pix0;
    #pragma unroll
    for (int mm = 0; mm < 4; ++mm)
        #pragma unroll
        for (int nt = 0; nt < 2; ++nt)
            #pragma unroll
            for (int r = 0; r < 8; ++r) {
                const int dd = mm * 16 + half * 8 + r;
                const int p  = wrow + nt * 16 + ln;
                const size_t idx = base + (size_t)dd * HWPIX + p;
                out[idx] = feature[idx] + acc2[mm][nt][r];
            }
}

// ---------------------------------------------------------------------------
extern "C" void kernel_launch(void* const* d_in, const int* in_sizes, int n_in,
                              void* d_out, int out_size, void* d_ws, size_t ws_size,
                              hipStream_t stream) {
    const float* feature = (const float*)d_in[0];
    const float* token   = (const float*)d_in[1];
    const float* Wv      = (const float*)d_in[2];
    const float* bv      = (const float*)d_in[3];
    const float* Wk      = (const float*)d_in[4];
    const float* bk      = (const float*)d_in[5];
    float*       out     = (float*)d_out;

    _Float16* vbuf = (_Float16*)d_ws;                       // 64*64*64 f16 = 512 KB
    _Float16* kbuf = vbuf + (size_t)NB * HEADS * DHEAD * LTOK; // another 512 KB

    proj_kv_kernel<<<NB * CCH, 64, 0, stream>>>(token, Wv, bv, Wk, bk, vbuf, kbuf);

    dim3 grid(HWPIX / PIXB, NB * HEADS);   // 64 x 64 blocks
    attn_main_kernel<<<grid, 256, 0, stream>>>(feature, kbuf, vbuf, out);
}
